// MultiHeadSelfAttention_9560597201502
// MI455X (gfx1250) — matrix operand order repaired and checked by reading, not compiled, not run on hardware
//
#include <hip/hip_runtime.h>
#include <hip/hip_bf16.h>

// ---------------------------------------------------------------------------
// MHSA forward for MI455X (gfx1250, wave32, WMMA f32_16x16x32_f16)
//   B=4, S=2048, D=1024, H=16, HD=64
// Pipeline:
//   [0] fp32 -> f16 converts (X, Wqkv, Wout)
//   [1] QKV GEMM  (f16 WMMA, fp32 accum, wave tile 32Mx64N); epilogue writes
//       Q*0.125 / K as [B,H,S,HD] f16 and V transposed as [B,H,HD,S] f16
//   [2] flash attention, causal, per-wave 16 query rows, 64-key blocks
//   [3] output projection GEMM -> fp32 d_out
// ---------------------------------------------------------------------------

typedef _Float16 v16h __attribute__((ext_vector_type(16)));
typedef _Float16 v8h  __attribute__((ext_vector_type(8)));
typedef float    v8f  __attribute__((ext_vector_type(8)));

#define BB 4
#define SS 2048
#define DD 1024
#define HH 16
#define HD 64

__device__ __forceinline__ v8f wmma_f16(v16h a, v16h b, v8f c) {
  // D = A(16x32 f16) * B(32x16 f16) + C(16x16 f32)
  return __builtin_amdgcn_wmma_f32_16x16x32_f16(
      /*neg_a=*/false, a, /*neg_b=*/false, b,
      /*c_mod=*/(short)0, c, /*reuse_a=*/false, /*reuse_b=*/false);
}

// A fragment (16x32, MxK): lane holds row M=lane&15; K halves
// [kb..kb+7] and [kb+16..kb+23], kb = (lane<16)?0:8.  Two 16B loads.
__device__ __forceinline__ v16h load_afrag(const _Float16* __restrict__ base,
                                           int row0, int ld, int k0) {
  int lane = threadIdx.x & 31;
  const _Float16* p = base + (size_t)(row0 + (lane & 15)) * ld + k0
                           + ((lane < 16) ? 0 : 8);
  v8h lo = *(const v8h*)(p);
  v8h hi = *(const v8h*)(p + 16);
  return __builtin_shufflevector(lo, hi, 0,1,2,3,4,5,6,7,8,9,10,11,12,13,14,15);
}

// B fragment (32x16, KxN) where B(k,n) = SRC[n0+n][k0+k] (SRC row-major,
// row stride ld): lane holds column n=lane&15; K halves
// [kb..kb+7] and [kb+16..kb+23], kb = (lane<16)?0:8.  Two 16B loads.
__device__ __forceinline__ v16h load_bfrag(const _Float16* __restrict__ base,
                                           int n0, int ld, int k0) {
  int lane = threadIdx.x & 31;
  const _Float16* p = base + (size_t)(n0 + (lane & 15)) * ld + k0
                           + ((lane < 16) ? 0 : 8);
  v8h lo = *(const v8h*)(p);
  v8h hi = *(const v8h*)(p + 16);
  return __builtin_shufflevector(lo, hi, 0,1,2,3,4,5,6,7,8,9,10,11,12,13,14,15);
}

// ------------------------- [0] fp32 -> f16 ---------------------------------
__global__ void f32_to_f16_kernel(const float* __restrict__ in,
                                  _Float16* __restrict__ out, int n) {
  int i = blockIdx.x * blockDim.x + threadIdx.x;
  if (i < n) out[i] = (_Float16)in[i];
}

// ------------------------- [1] QKV GEMM ------------------------------------
// C[8192,3072] = X[8192,1024] * Wqkv^T ; block = 4 waves, tile 128(M)x64(N),
// each wave: 32(M)x64(N) = 8 accumulators (B frags reused across 2 A frags).
__global__ void qkv_gemm_kernel(const _Float16* __restrict__ X,
                                const _Float16* __restrict__ W,
                                _Float16* __restrict__ Q,
                                _Float16* __restrict__ Kt,
                                _Float16* __restrict__ Vt) {
  int wave = threadIdx.x >> 5;
  int lane = threadIdx.x & 31;
  int m0 = blockIdx.x * 128 + wave * 32;
  int n0 = blockIdx.y * 64;

  v8f acc[2][4];
#pragma unroll
  for (int mi = 0; mi < 2; ++mi)
#pragma unroll
    for (int t = 0; t < 4; ++t) acc[mi][t] = (v8f)0.0f;

  for (int k0 = 0; k0 < DD; k0 += 32) {
    // prefetch two k-tiles ahead (wrapped, branch-free; global_prefetch_b8)
    int kpf = (k0 + 64) & (DD - 1);
    __builtin_prefetch(X + (size_t)(m0 + lane) * DD + kpf, 0, 3);
    __builtin_prefetch(W + (size_t)(n0 + lane) * DD + kpf, 0, 3);
    __builtin_prefetch(W + (size_t)(n0 + 32 + lane) * DD + kpf, 0, 3);

    v16h a0 = load_afrag(X, m0, DD, k0);
    v16h a1 = load_afrag(X, m0 + 16, DD, k0);
#pragma unroll
    for (int t = 0; t < 4; ++t) {
      v16h b = load_bfrag(W, n0 + t * 16, DD, k0);
      acc[0][t] = wmma_f16(a0, b, acc[0][t]);
      acc[1][t] = wmma_f16(a1, b, acc[1][t]);
    }
  }

  int which = n0 >> 10;  // 0=Q 1=K 2=V (64-col tile never crosses a D block)
  int halfsel = (lane < 16) ? 0 : 8;
#pragma unroll
  for (int mi = 0; mi < 2; ++mi) {
#pragma unroll
    for (int t = 0; t < 4; ++t) {
#pragma unroll
      for (int i = 0; i < 8; ++i) {
        int m = m0 + mi * 16 + i + halfsel;
        int n = n0 + t * 16 + (lane & 15);
        int d = n & (DD - 1);
        int h = d >> 6;
        int hd = d & (HD - 1);
        int b = m >> 11;
        int s = m & (SS - 1);
        float v = acc[mi][t][i];
        size_t bh = (size_t)(b * HH + h);
        if (which == 0) {
          Q[(bh * SS + s) * HD + hd] = (_Float16)(v * 0.125f);  // fold HD^-0.5
        } else if (which == 1) {
          Kt[(bh * SS + s) * HD + hd] = (_Float16)v;
        } else {
          Vt[(bh * HD + hd) * SS + s] = (_Float16)v;            // V transposed
        }
      }
    }
  }
}

// ------------------------- [2] flash attention -----------------------------
// grid = B*H*(S/64) blocks, 4 waves/block, wave owns 16 query rows,
// iterates 64-key blocks up to the causal diagonal.
__global__ void flash_attn_kernel(const _Float16* __restrict__ Q,
                                  const _Float16* __restrict__ K,
                                  const _Float16* __restrict__ Vt,
                                  _Float16* __restrict__ Oa) {
  __shared__ _Float16 ldsP[4][16 * 64];  // per-wave P tile (C->A relayout)

  int wave = threadIdx.x >> 5;
  int lane = threadIdx.x & 31;
  int qtile = blockIdx.x & 31;
  int bh = blockIdx.x >> 5;
  int b = bh >> 4;
  int h = bh & 15;

  const _Float16* Qh = Q + (size_t)bh * SS * HD;
  const _Float16* Kh = K + (size_t)bh * SS * HD;
  const _Float16* Vh = Vt + (size_t)bh * HD * SS;

  int q_base = qtile * 64 + wave * 16;
  int halfsel = (lane < 16) ? 0 : 8;

  v16h qa0 = load_afrag(Qh, q_base, HD, 0);
  v16h qa1 = load_afrag(Qh, q_base, HD, 32);

  v8f o[4];
#pragma unroll
  for (int t = 0; t < 4; ++t) o[t] = (v8f)0.0f;
  float rm[8], rl[8];
#pragma unroll
  for (int i = 0; i < 8; ++i) { rm[i] = -__builtin_inff(); rl[i] = 0.0f; }

  int q_max = q_base + 15;
  for (int j = 0; j <= q_max; j += 64) {   // causal: only blocks <= diagonal
    // warm L2/L0 for the next key block (addresses stay inside d_ws)
    __builtin_prefetch(Kh + (size_t)(j + 64 + lane) * HD, 0, 3);
    __builtin_prefetch(Vh + (size_t)(lane)*SS + j + 64, 0, 3);
    __builtin_prefetch(Vh + (size_t)(32 + lane) * SS + j + 64, 0, 3);

    v8f s[4];
#pragma unroll
    for (int u = 0; u < 4; ++u) {
      s[u] = (v8f)0.0f;
      s[u] = wmma_f16(qa0, load_bfrag(Kh, j + u * 16, HD, 0), s[u]);
      s[u] = wmma_f16(qa1, load_bfrag(Kh, j + u * 16, HD, 32), s[u]);
    }

    int kc = j + (lane & 15);
#pragma unroll
    for (int i = 0; i < 8; ++i) {
      int qrow = q_base + i + halfsel;
      float v[4];
#pragma unroll
      for (int u = 0; u < 4; ++u)
        v[u] = (kc + u * 16 > qrow) ? -__builtin_inff() : s[u][i];
      float mx = fmaxf(fmaxf(v[0], v[1]), fmaxf(v[2], v[3]));
      mx = fmaxf(mx, __shfl_xor(mx, 1));
      mx = fmaxf(mx, __shfl_xor(mx, 2));
      mx = fmaxf(mx, __shfl_xor(mx, 4));
      mx = fmaxf(mx, __shfl_xor(mx, 8));   // reduces within 16-lane half
      float mnew = fmaxf(rm[i], mx);
      float alpha = __expf(rm[i] - mnew);
      float p[4];
      float rs = 0.0f;
#pragma unroll
      for (int u = 0; u < 4; ++u) { p[u] = __expf(v[u] - mnew); rs += p[u]; }
      rs += __shfl_xor(rs, 1);
      rs += __shfl_xor(rs, 2);
      rs += __shfl_xor(rs, 4);
      rs += __shfl_xor(rs, 8);
      rl[i] = rl[i] * alpha + rs;
      rm[i] = mnew;
      o[0][i] *= alpha; o[1][i] *= alpha; o[2][i] *= alpha; o[3][i] *= alpha;
      int mrow = i + halfsel;
#pragma unroll
      for (int u = 0; u < 4; ++u)
        ldsP[wave][mrow * 64 + u * 16 + (lane & 15)] = (_Float16)p[u];
    }
    // wave-synchronous (DS in-order per wave): read back P in A layout
    v16h pa0, pa1;
    {
      int m = lane & 15;
      int kb = (lane < 16) ? 0 : 8;
      v8h lo0 = *(const v8h*)&ldsP[wave][m * 64 + kb];
      v8h hi0 = *(const v8h*)&ldsP[wave][m * 64 + kb + 16];
      v8h lo1 = *(const v8h*)&ldsP[wave][m * 64 + 32 + kb];
      v8h hi1 = *(const v8h*)&ldsP[wave][m * 64 + 32 + kb + 16];
      pa0 = __builtin_shufflevector(lo0, hi0,
                                    0,1,2,3,4,5,6,7,8,9,10,11,12,13,14,15);
      pa1 = __builtin_shufflevector(lo1, hi1,
                                    0,1,2,3,4,5,6,7,8,9,10,11,12,13,14,15);
    }
#pragma unroll
    for (int t = 0; t < 4; ++t) {
      o[t] = wmma_f16(pa0, load_bfrag(Vh, t * 16, SS, j), o[t]);
      o[t] = wmma_f16(pa1, load_bfrag(Vh, t * 16, SS, j + 32), o[t]);
    }
  }

  // epilogue: O / l, store f16 to [B,S,D] for the output projection
#pragma unroll
  for (int i = 0; i < 8; ++i) {
    float inv = 1.0f / rl[i];
    int qrow = q_base + i + halfsel;
    size_t row = (size_t)b * SS + qrow;
#pragma unroll
    for (int t = 0; t < 4; ++t) {
      int col = h * HD + t * 16 + (lane & 15);
      Oa[row * DD + col] = (_Float16)(o[t][i] * inv);
    }
  }
}

// ------------------------- [3] output projection ---------------------------
__global__ void out_gemm_kernel(const _Float16* __restrict__ A,
                                const _Float16* __restrict__ W,
                                float* __restrict__ out) {
  int wave = threadIdx.x >> 5;
  int lane = threadIdx.x & 31;
  int m0 = blockIdx.x * 128 + wave * 32;
  int n0 = blockIdx.y * 64;

  v8f acc[2][4];
#pragma unroll
  for (int mi = 0; mi < 2; ++mi)
#pragma unroll
    for (int t = 0; t < 4; ++t) acc[mi][t] = (v8f)0.0f;

  for (int k0 = 0; k0 < DD; k0 += 32) {
    int kpf = (k0 + 64) & (DD - 1);
    __builtin_prefetch(A + (size_t)(m0 + lane) * DD + kpf, 0, 3);
    __builtin_prefetch(W + (size_t)(n0 + lane) * DD + kpf, 0, 3);
    __builtin_prefetch(W + (size_t)(n0 + 32 + lane) * DD + kpf, 0, 3);

    v16h a0 = load_afrag(A, m0, DD, k0);
    v16h a1 = load_afrag(A, m0 + 16, DD, k0);
#pragma unroll
    for (int t = 0; t < 4; ++t) {
      v16h b = load_bfrag(W, n0 + t * 16, DD, k0);
      acc[0][t] = wmma_f16(a0, b, acc[0][t]);
      acc[1][t] = wmma_f16(a1, b, acc[1][t]);
    }
  }

  int halfsel = (lane < 16) ? 0 : 8;
#pragma unroll
  for (int mi = 0; mi < 2; ++mi) {
#pragma unroll
    for (int t = 0; t < 4; ++t) {
#pragma unroll
      for (int i = 0; i < 8; ++i) {
        int m = m0 + mi * 16 + i + halfsel;
        int n = n0 + t * 16 + (lane & 15);
        out[(size_t)m * DD + n] = acc[mi][t][i];
      }
    }
  }
}

// ---------------------------------------------------------------------------
extern "C" void kernel_launch(void* const* d_in, const int* in_sizes, int n_in,
                              void* d_out, int out_size, void* d_ws,
                              size_t ws_size, hipStream_t stream) {
  (void)in_sizes; (void)n_in; (void)out_size; (void)ws_size;
  const float* X    = (const float*)d_in[0];  // [B,S,D]
  const float* Wqkv = (const float*)d_in[1];  // [3D,D]
  const float* Wout = (const float*)d_in[2];  // [D,D]
  // d_in[3] attention_mask (all true), d_in[4] causal mask (triu) -> analytic

  char* ws = (char*)d_ws;
  const size_t MB = 1ull << 20;
  _Float16* Xh    = (_Float16*)(ws + 0 * MB);   // 16 MB  [8192,1024]
  _Float16* Wqkvh = (_Float16*)(ws + 16 * MB);  //  6 MB  [3072,1024]
  _Float16* Wouth = (_Float16*)(ws + 22 * MB);  //  2 MB  [1024,1024]
  _Float16* Qb    = (_Float16*)(ws + 24 * MB);  // 16 MB  [B,H,S,HD] (scaled)
  _Float16* Kb    = (_Float16*)(ws + 40 * MB);  // 16 MB  [B,H,S,HD]
  _Float16* Vb    = (_Float16*)(ws + 56 * MB);  // 16 MB  [B,H,HD,S] (transposed)
  _Float16* Ab    = (_Float16*)(ws + 72 * MB);  // 16 MB  [B,S,D]

  const int nX = BB * SS * DD;        // 8388608
  const int nWq = 3 * DD * DD;        // 3145728
  const int nWo = DD * DD;            // 1048576
  f32_to_f16_kernel<<<(nX + 255) / 256, 256, 0, stream>>>(X, Xh, nX);
  f32_to_f16_kernel<<<(nWq + 255) / 256, 256, 0, stream>>>(Wqkv, Wqkvh, nWq);
  f32_to_f16_kernel<<<(nWo + 255) / 256, 256, 0, stream>>>(Wout, Wouth, nWo);

  // QKV GEMM: M=8192, N=3072, 128x64 block tiles
  qkv_gemm_kernel<<<dim3(BB * SS / 128, 3 * DD / 64), 128, 0, stream>>>(
      Xh, Wqkvh, Qb, Kb, Vb);

  // Flash attention: B*H*(S/64) blocks
  flash_attn_kernel<<<BB * HH * (SS / 64), 128, 0, stream>>>(Qb, Kb, Vb, Ab);

  // Output projection: M=8192, N=1024, 128x64 block tiles
  out_gemm_kernel<<<dim3(BB * SS / 128, DD / 64), 128, 0, stream>>>(
      Ab, Wouth, (float*)d_out);
}
